// UpProj_Block_40733469835401
// MI455X (gfx1250) — matrix operand order repaired and checked by reading, not compiled, not run on hardware
//
#include <hip/hip_runtime.h>

typedef __attribute__((ext_vector_type(16))) __bf16 v16bf;
typedef __attribute__((ext_vector_type(8)))  float  v8f;

// ---------------- workspace layout (bytes) ----------------
// xbp  : bf16 NHWC x, zero-padded halo (16,30,34,512)   16,711,680
// wp1  : bf16 packed w1*s1 (25 taps)                     6,553,600
// wpsc : bf16 packed wsc*ss (25 taps)                    6,553,600
// wp2  : bf16 packed w2*s2 (9 taps)                      1,179,648
// out1p: bf16 NHWC, zero-padded halo (16,58,66,256)     31,309,824
// scb  : bf16 NCHW shortcut (16,256,56,56)              25,690,112
// bnp  : f32 [s1,b1f,s2,b2f,ss,bsf]                          6,144
static constexpr size_t XB_OFF  = 0;
static constexpr size_t WP1_OFF = 16711680;
static constexpr size_t WPS_OFF = 23265280;
static constexpr size_t WP2_OFF = 29818880;
static constexpr size_t O1_OFF  = 30998528;
static constexpr size_t SC_OFF  = 62308352;
static constexpr size_t BN_OFF  = 87998464;

// ---------------- prep kernels ----------------
__global__ void bn_fold(const float* g1, const float* b1, const float* m1, const float* v1,
                        const float* g2, const float* b2, const float* m2, const float* v2,
                        const float* gs, const float* bs, const float* ms, const float* vs,
                        float* bnp) {
  int c = threadIdx.x; // 256 threads
  float s1 = g1[c] * rsqrtf(v1[c] + 1e-5f);
  bnp[c] = s1;           bnp[256 + c]  = b1[c] - m1[c] * s1;
  float s2 = g2[c] * rsqrtf(v2[c] + 1e-5f);
  bnp[512 + c] = s2;     bnp[768 + c]  = b2[c] - m2[c] * s2;
  float ss = gs[c] * rsqrtf(vs[c] + 1e-5f);
  bnp[1024 + c] = ss;    bnp[1280 + c] = bs[c] - ms[c] * ss;
}

// x: f32 NCHW (16,512,28,28) -> bf16 padded NHWC (16,30,34,512), halo = 0
__global__ void convert_x(const float* __restrict__ x, __bf16* __restrict__ xbp) {
  int tid = blockIdx.x * 256 + threadIdx.x;   // 8,355,840 total
  int c = tid & 511;
  int pos = tid >> 9;
  int jp = pos % 34; pos /= 34;
  int ip = pos % 30;
  int b  = pos / 30;
  int i = ip - 1, j = jp - 1;
  float v = 0.0f;
  if (i >= 0 && i < 28 && j >= 0 && j < 28)
    v = x[(((size_t)b * 512 + c) * 28 + i) * 28 + j];
  xbp[tid] = (__bf16)v;
}

// zero 16 bytes per thread (used to clear padded out1 buffer)
__global__ void zero16(uint4* __restrict__ p) {
  uint4 z; z.x = 0; z.y = 0; z.z = 0; z.w = 0;
  p[blockIdx.x * 256 + threadIdx.x] = z;
}

// pack OIHW f32 weights (scale folded) into WMMA A-fragment order:
// index = ((tap*16 + mtile)*nks + ks)*512 + lane*16 + h
// lane<16: h0..15 => K=ks*32+0..15 ; lane>=16 shifted by +16
__global__ void pack_w(const float* __restrict__ w, const float* __restrict__ scale,
                       __bf16* __restrict__ wp, int Cin, int kdim, int ntotal) {
  int tid = blockIdx.x * 256 + threadIdx.x;
  if (tid >= ntotal) return;
  int h    = tid & 15;
  int lane = (tid >> 4) & 31;
  int rest = tid >> 9;
  int nks  = Cin >> 5;
  int ks   = rest % nks; rest /= nks;
  int mt   = rest & 15;
  int tap  = rest >> 4;
  int sel  = lane >> 4;
  int oc   = mt * 16 + (lane & 15);
  int kk   = sel * 16 + h;
  int ic   = ks * 32 + kk;
  int kh   = tap / kdim, kw = tap % kdim;
  float val = w[(((size_t)oc * Cin + ic) * kdim + kh) * kdim + kw] * scale[oc];
  wp[tid] = (__bf16)val;
}

// ---------------- stage 1: conv1 (5x5 transposed) + shortcut conv ----------------
// grid.x = 16*56*2 ; block = 256 (8 waves: conv{0,1} x 4 oc-groups of 64)
// each wave: 4 oc-tiles x 2 column-tiles (z = n and n+16), unconditional B loads (halo)
__global__ __launch_bounds__(256) void stage1(
    const __bf16* __restrict__ xbp, const __bf16* __restrict__ wp1,
    const __bf16* __restrict__ wpsc, const float* __restrict__ bnp,
    __bf16* __restrict__ out1p, __bf16* __restrict__ scb) {
  const int bid = blockIdx.x;
  const int pw  = bid & 1;                 // column parity
  const int oh  = (bid >> 1) % 56;         // output row
  const int b   = bid / 112;
  const int lane = threadIdx.x & 31;
  const int wave = threadIdx.x >> 5;
  const int conv = wave >> 2;              // 0 = conv1, 1 = shortcut
  const int mg   = wave & 3;               // oc group of 64
  const int n    = lane & 15;
  const int sel  = lane >> 4;

  const __bf16* wp   = conv ? wpsc : wp1;
  const float*  bias = bnp + (conv ? 1280 : 256);

  v8f acc[4][2];
#pragma unroll
  for (int m = 0; m < 4; ++m)
#pragma unroll
    for (int r = 0; r < 8; ++r) {
      float bv = bias[mg * 64 + m * 16 + r + sel * 8];
      acc[m][0][r] = bv;
      acc[m][1][r] = bv;
    }

  for (int kh = 0; kh < 5; ++kh) {
    if ((oh + kh) & 1) continue;           // tap lands on inserted zero row
    const int i = (oh - 2 + kh) >> 1;      // -1..28, covered by halo
    const size_t rowoff = (size_t)(b * 30 + i + 1) * 34;
    for (int kw = 0; kw < 5; ++kw) {
      if ((pw + kw) & 1) continue;         // tap lands on inserted zero column
      const int d = (pw - 2 + kw) >> 1;    // -1..1
      const int tap = kh * 5 + kw;
      const __bf16* bp0 = xbp + (rowoff + (n + d + 1)) * 512 + sel * 16;
      const __bf16* bp1 = bp0 + 16 * 512;
      const __bf16* ab  = wp + (size_t)(tap * 16 + mg * 4) * 16 * 512 + lane * 16;
#pragma unroll 2
      for (int ks = 0; ks < 16; ++ks) {
        const v16bf bf0 = *(const v16bf*)(bp0 + ks * 32);
        const v16bf bf1 = *(const v16bf*)(bp1 + ks * 32);
#pragma unroll
        for (int m = 0; m < 4; ++m) {
          const v16bf af = *(const v16bf*)(ab + (size_t)(m * 16 + ks) * 512);
          acc[m][0] = __builtin_amdgcn_wmma_f32_16x16x32_bf16(
              false, af, false, bf0, (short)0, acc[m][0], false, false);
          acc[m][1] = __builtin_amdgcn_wmma_f32_16x16x32_bf16(
              false, af, false, bf1, (short)0, acc[m][1], false, false);
        }
      }
    }
  }

  const int ow0 = 2 * n + pw;              // z = n       (always valid)
  const int ow1 = 2 * (n + 16) + pw;       // z = n + 16  (valid iff n < 12)
  if (conv == 0) {
    const size_t rb = (size_t)(b * 58 + oh + 1) * 66 * 256;
#pragma unroll
    for (int m = 0; m < 4; ++m)
#pragma unroll
      for (int r = 0; r < 8; ++r) {
        int oc = mg * 64 + m * 16 + r + sel * 8;
        float v0 = acc[m][0][r]; v0 = v0 > 0.0f ? v0 : 0.0f;
        out1p[rb + (size_t)(ow0 + 1) * 256 + oc] = (__bf16)v0;
        if (n < 12) {
          float v1 = acc[m][1][r]; v1 = v1 > 0.0f ? v1 : 0.0f;
          out1p[rb + (size_t)(ow1 + 1) * 256 + oc] = (__bf16)v1;
        }
      }
  } else {
#pragma unroll
    for (int m = 0; m < 4; ++m)
#pragma unroll
      for (int r = 0; r < 8; ++r) {
        int oc = mg * 64 + m * 16 + r + sel * 8;
        const size_t cb = ((size_t)(b * 256 + oc) * 56 + oh) * 56;
        scb[cb + ow0] = (__bf16)acc[m][0][r];
        if (n < 12) scb[cb + ow1] = (__bf16)acc[m][1][r];
      }
  }
}

// ---------------- stage 2: dense 3x3 conv + bias + shortcut add + ReLU ----------------
// grid.x = 16*56*2 ; block = 128 (4 waves x 64 oc), each wave 4 oc-tiles x 2 ow-tiles
__global__ __launch_bounds__(128) void stage2(
    const __bf16* __restrict__ out1p, const __bf16* __restrict__ wp2,
    const __bf16* __restrict__ scb, const float* __restrict__ bnp,
    float* __restrict__ out) {
  const int bid = blockIdx.x;
  const int wtp = bid & 1;                 // ow tile pair: {0,1} or {2,3}
  const int oh  = (bid >> 1) % 56;
  const int b   = bid / 112;
  const int lane = threadIdx.x & 31;
  const int wave = threadIdx.x >> 5;       // oc group of 64
  const int n    = lane & 15;
  const int sel  = lane >> 4;
  const int ow0  = wtp * 32 + n;           // always < 56
  const int ow1  = ow0 + 16;               // valid iff < 56
  const bool cv1 = (ow1 < 56);
  const int owc1 = cv1 ? ow1 : 55;
  const float* b2f = bnp + 768;

  v8f acc[4][2];
#pragma unroll
  for (int m = 0; m < 4; ++m)
#pragma unroll
    for (int r = 0; r < 8; ++r) {
      int oc = wave * 64 + m * 16 + r + sel * 8;
      const size_t cb = ((size_t)(b * 256 + oc) * 56 + oh) * 56;
      acc[m][0][r] = b2f[oc] + (float)scb[cb + ow0];
      acc[m][1][r] = b2f[oc] + (float)scb[cb + owc1];
    }

  for (int kh = 0; kh < 3; ++kh) {
    const int ih = oh - 1 + kh;            // -1..56, covered by halo
    const size_t rowoff = (size_t)(b * 58 + ih + 1) * 66;
    for (int kw = 0; kw < 3; ++kw) {
      const int tap = kh * 3 + kw;
      const __bf16* bp0 = out1p + (rowoff + (ow0 + kw)) * 256 + sel * 16;
      const __bf16* bp1 = bp0 + 16 * 256;
      const __bf16* ab  = wp2 + (size_t)(tap * 16 + wave * 4) * 8 * 512 + lane * 16;
#pragma unroll 2
      for (int ks = 0; ks < 8; ++ks) {
        const v16bf bf0 = *(const v16bf*)(bp0 + ks * 32);
        const v16bf bf1 = *(const v16bf*)(bp1 + ks * 32);
#pragma unroll
        for (int m = 0; m < 4; ++m) {
          const v16bf af = *(const v16bf*)(ab + (size_t)(m * 8 + ks) * 512);
          acc[m][0] = __builtin_amdgcn_wmma_f32_16x16x32_bf16(
              false, af, false, bf0, (short)0, acc[m][0], false, false);
          acc[m][1] = __builtin_amdgcn_wmma_f32_16x16x32_bf16(
              false, af, false, bf1, (short)0, acc[m][1], false, false);
        }
      }
    }
  }

#pragma unroll
  for (int m = 0; m < 4; ++m)
#pragma unroll
    for (int r = 0; r < 8; ++r) {
      int oc = wave * 64 + m * 16 + r + sel * 8;
      const size_t ob = ((size_t)(b * 256 + oc) * 56 + oh) * 56;
      float v0 = acc[m][0][r]; v0 = v0 > 0.0f ? v0 : 0.0f;
      out[ob + ow0] = v0;
      if (cv1) {
        float v1 = acc[m][1][r]; v1 = v1 > 0.0f ? v1 : 0.0f;
        out[ob + ow1] = v1;
      }
    }
}

// ---------------- host launch ----------------
extern "C" void kernel_launch(void* const* d_in, const int* in_sizes, int n_in,
                              void* d_out, int out_size, void* d_ws, size_t ws_size,
                              hipStream_t stream) {
  const float* x   = (const float*)d_in[0];
  const float* w1  = (const float*)d_in[1];
  const float* g1  = (const float*)d_in[2];
  const float* b1  = (const float*)d_in[3];
  const float* m1  = (const float*)d_in[4];
  const float* v1  = (const float*)d_in[5];
  const float* w2  = (const float*)d_in[6];
  const float* g2  = (const float*)d_in[7];
  const float* b2  = (const float*)d_in[8];
  const float* m2  = (const float*)d_in[9];
  const float* v2  = (const float*)d_in[10];
  const float* wsc = (const float*)d_in[11];
  const float* gs  = (const float*)d_in[12];
  const float* bs  = (const float*)d_in[13];
  const float* ms  = (const float*)d_in[14];
  const float* vs  = (const float*)d_in[15];

  char* ws = (char*)d_ws;
  __bf16* xbp   = (__bf16*)(ws + XB_OFF);
  __bf16* wp1   = (__bf16*)(ws + WP1_OFF);
  __bf16* wpsc  = (__bf16*)(ws + WPS_OFF);
  __bf16* wp2   = (__bf16*)(ws + WP2_OFF);
  __bf16* out1p = (__bf16*)(ws + O1_OFF);
  __bf16* scb   = (__bf16*)(ws + SC_OFF);
  float*  bnp   = (float*)(ws + BN_OFF);

  bn_fold<<<1, 256, 0, stream>>>(g1, b1, m1, v1, g2, b2, m2, v2, gs, bs, ms, vs, bnp);
  convert_x<<<32640, 256, 0, stream>>>(x, xbp);
  pack_w<<<12800, 256, 0, stream>>>(w1,  bnp,        wp1,  512, 5, 3276800);
  pack_w<<<12800, 256, 0, stream>>>(wsc, bnp + 1024, wpsc, 512, 5, 3276800);
  pack_w<<<2304,  256, 0, stream>>>(w2,  bnp + 512,  wp2,  256, 3, 589824);
  zero16<<<7644, 256, 0, stream>>>((uint4*)out1p);   // clear padded out1 (incl. halo)
  stage1<<<1792, 256, 0, stream>>>(xbp, wp1, wpsc, bnp, out1p, scb);
  stage2<<<1792, 128, 0, stream>>>(out1p, wp2, scb, bnp, (float*)d_out);
}
